// TimeSeriesTransformerDecoder_77300821393605
// MI455X (gfx1250) — compile-verified, matrix operand order checked by reading, not run on hardware
//
#include <hip/hip_runtime.h>
#include <hip/hip_bf16.h>

// ---------------------------------------------------------------------------
// TimeSeriesTransformerDecoder for MI455X (gfx1250, wave32, WMMA).
//
// Key restructurings vs the reference (mathematically exact):
//  * causal mask => incremental decode with per-layer K/V caches (f16)
//  * cross-attention has exactly 1 key => softmax==1 => its contribution is
//    a per-layer constant  ca_add[l] = (mem@Wv.T+bv)@Wo.T+bo, precomputed.
// GEMMs run on v_wmma_f32_16x16x32_f16 (f16 in, f32 accumulate); each wave
// computes a 16(M)x64(N) strip, reusing one A fragment across 4 WMMAs.
// ---------------------------------------------------------------------------

#define DEV static __device__ __forceinline__

typedef __attribute__((ext_vector_type(16))) _Float16 v16h;
typedef __attribute__((ext_vector_type(8)))  _Float16 v8h;
typedef __attribute__((ext_vector_type(8)))  float    v8f;

namespace cfg {
constexpr int B  = 256;
constexpr int D  = 512;
constexpr int H  = 8;
constexpr int HD = 64;
constexpr int L  = 4;
constexpr int FF = 2048;
constexpr int P  = 24;
}

DEV float gelu_f(float x) {
    return 0.5f * x * (1.0f + erff(x * 0.70710678118654752440f));
}

// Load one 16x32 (rows x K) f16 WMMA fragment.  Both A[M,K] and W[N,K] are
// row-major with K contiguous, so A and B fragments use the same pattern:
//   lane&15 selects the row; lanes>=16 take the K+8 / K+24 element groups.
DEV v16h load_frag16(const _Float16* __restrict__ base, int ld, int kb, int lane) {
    const int row  = lane & 15;
    const int koff = (lane >> 4) * 8;
    const _Float16* p = base + (size_t)row * ld + kb + koff;
    union { v16h v; v8h h[2]; } u;
    u.h[0] = *(const v8h*)(p);          // K = kb+koff   .. +7
    u.h[1] = *(const v8h*)(p + 16);     // K = kb+16+koff.. +7
    return u.v;
}

// C = act(A[M,K] @ W[N,K]^T + bias);  f32 output, optional f16 mirror.
// One wave -> 16(M) x 64(N) tile.  ACT: 0 = none, 1 = exact GELU.
template <int ACT>
__global__ void gemm_wmma_kernel(const _Float16* __restrict__ A,
                                 const _Float16* __restrict__ W,
                                 const float*    __restrict__ bias,
                                 float*          __restrict__ Cf,
                                 _Float16*       __restrict__ Ch,
                                 int M, int N, int K)
{
    const int lane   = threadIdx.x & 31;
    const int wid    = (blockIdx.x * blockDim.x + threadIdx.x) >> 5;
    const int wavesN = N >> 6;                 // 64-wide N strips
    const int wavesM = M >> 4;
    if (wid >= wavesM * wavesN) return;        // wave-uniform: EXEC all-1 below
    const int wm = wid / wavesN;
    const int wn = wid % wavesN;

    const _Float16* Abase = A + (size_t)wm * 16 * K;
    v8f acc[4] = {};

    for (int kb = 0; kb < K; kb += 32) {
        const v16h a = load_frag16(Abase, K, kb, lane);
#pragma unroll
        for (int j = 0; j < 4; ++j) {
            const _Float16* Wbase = W + (size_t)(wn * 64 + j * 16) * K;
            const v16h b = load_frag16(Wbase, K, kb, lane);
            acc[j] = __builtin_amdgcn_wmma_f32_16x16x32_f16(
                false, a, false, b, (short)0, acc[j], false, false);
        }
    }

    const int n0 = wn * 64 + (lane & 15);
    const int m0 = wm * 16 + ((lane >> 4) << 3);
#pragma unroll
    for (int j = 0; j < 4; ++j) {
        const int n  = n0 + j * 16;
        const float bv = bias ? bias[n] : 0.0f;
#pragma unroll
        for (int v = 0; v < 8; ++v) {
            const int m = m0 + v;
            float val = acc[j][v] + bv;
            if (ACT == 1) val = gelu_f(val);
            Cf[(size_t)m * N + n] = val;
            if (Ch) Ch[(size_t)m * N + n] = (_Float16)val;
        }
    }
}

__global__ void cvt_f32_to_f16_kernel(const float* __restrict__ in,
                                      _Float16* __restrict__ out, int n)
{
    const int i = blockIdx.x * blockDim.x + threadIdx.x;
    if (i < n) out[i] = (_Float16)in[i];
}

__global__ void init_vals_kernel(float* __restrict__ vals,
                                 const float* __restrict__ start_token)
{
    vals[threadIdx.x] = start_token[0];           // vals[0][b]
}

// x[b,d] = vals_t[b]*ve_W[d] + ve_b[d] + pe(t,d)
__global__ void embed_kernel(const float* __restrict__ vals_t,
                             const float* __restrict__ veW,
                             const float* __restrict__ veb,
                             float* __restrict__ x, _Float16* __restrict__ xh, int t)
{
    using namespace cfg;
    const int b = blockIdx.x, d = threadIdx.x;    // blockDim = 512
    const int i2 = d & ~1;
    const float freq = __expf(-(float)i2 * (9.210340371976184f / (float)D));
    const float ang  = (float)t * freq;
    const float pe   = (d & 1) ? __cosf(ang) : __sinf(ang);
    const float v = vals_t[b] * veW[d] + veb[d] + pe;
    x[(size_t)b * D + d]  = v;
    xh[(size_t)b * D + d] = (_Float16)v;
}

// Append K/V for position t into the layer's f16 caches ([P][B][D]).
__global__ void kv_append_kernel(const float* __restrict__ qkv,
                                 _Float16* __restrict__ kc,
                                 _Float16* __restrict__ vc, int t)
{
    using namespace cfg;
    const int i = blockIdx.x * blockDim.x + threadIdx.x;   // B*D
    if (i >= B * D) return;
    const int b = i >> 9, d = i & (D - 1);
    const size_t dst = ((size_t)t * B + b) * D + d;
    kc[dst] = (_Float16)qkv[(size_t)b * (3 * D) + D + d];
    vc[dst] = (_Float16)qkv[(size_t)b * (3 * D) + 2 * D + d];
}

// Causal self-attention for the single new token.  One wave per (b,h);
// lane l owns head dims {l, l+32}.  Keys/values: positions 0..t from cache.
__global__ void attn_kernel(const float* __restrict__ qkv,
                            const _Float16* __restrict__ kc,
                            const _Float16* __restrict__ vc,
                            _Float16* __restrict__ o, int t)
{
    using namespace cfg;
    const int bh   = blockIdx.x;               // B*H waves
    const int b    = bh >> 3, h = bh & (H - 1);
    const int lane = threadIdx.x;              // 32
    const int d0   = h * HD + lane, d1 = d0 + 32;

    const float q0 = qkv[(size_t)b * (3 * D) + d0];
    const float q1 = qkv[(size_t)b * (3 * D) + d1];

    float sc[P];
#pragma unroll
    for (int p = 0; p < P; ++p) {
        if (p <= t) {
            const _Float16* kp = kc + ((size_t)p * B + b) * D;
            float part = q0 * (float)kp[d0] + q1 * (float)kp[d1];
#pragma unroll
            for (int m = 16; m > 0; m >>= 1) part += __shfl_xor(part, m, 32);
            sc[p] = part * 0.125f;             // 1/sqrt(HD)
        } else sc[p] = -1e30f;
    }
    float mx = -1e30f;
#pragma unroll
    for (int p = 0; p < P; ++p) if (p <= t) mx = fmaxf(mx, sc[p]);
    float sum = 0.0f;
#pragma unroll
    for (int p = 0; p < P; ++p) { sc[p] = (p <= t) ? __expf(sc[p] - mx) : 0.0f; sum += sc[p]; }
    const float inv = 1.0f / sum;
    float a0 = 0.0f, a1 = 0.0f;
#pragma unroll
    for (int p = 0; p < P; ++p) {
        if (p <= t) {
            const _Float16* vp = vc + ((size_t)p * B + b) * D;
            const float w = sc[p] * inv;
            a0 += w * (float)vp[d0];
            a1 += w * (float)vp[d1];
        }
    }
    o[(size_t)b * D + d0] = (_Float16)a0;
    o[(size_t)b * D + d1] = (_Float16)a1;
}

// x = LayerNorm(x + r) * w + b ; refresh f16 mirror.  One block (256) per row.
__global__ void add_ln_kernel(float* __restrict__ x, const float* __restrict__ r,
                              const float* __restrict__ w, const float* __restrict__ bb,
                              _Float16* __restrict__ xh)
{
    using namespace cfg;
    const int b = blockIdx.x, tid = threadIdx.x;   // 256
    __shared__ float red[256];
    const size_t base = (size_t)b * D;
    const float v0 = x[base + tid]       + r[base + tid];
    const float v1 = x[base + tid + 256] + r[base + tid + 256];
    red[tid] = v0 + v1;
    __syncthreads();
    for (int s = 128; s > 0; s >>= 1) { if (tid < s) red[tid] += red[tid + s]; __syncthreads(); }
    const float mean = red[0] * (1.0f / (float)D);
    __syncthreads();
    const float dv0 = v0 - mean, dv1 = v1 - mean;
    red[tid] = dv0 * dv0 + dv1 * dv1;
    __syncthreads();
    for (int s = 128; s > 0; s >>= 1) { if (tid < s) red[tid] += red[tid + s]; __syncthreads(); }
    const float inv = rsqrtf(red[0] * (1.0f / (float)D) + 1e-5f);
    const float o0 = dv0 * inv * w[tid]       + bb[tid];
    const float o1 = dv1 * inv * w[tid + 256] + bb[tid + 256];
    x[base + tid]        = o0;  x[base + tid + 256]  = o1;
    xh[base + tid]       = (_Float16)o0;
    xh[base + tid + 256] = (_Float16)o1;
}

// nv = h1 @ o2_w.T + o2_b  (D/2 -> 1); write output[b,t] and vals[t+1][b].
__global__ void head2_kernel(const float* __restrict__ h1, const float* __restrict__ o2w,
                             const float* __restrict__ o2b, float* __restrict__ out,
                             float* __restrict__ vals, int t)
{
    using namespace cfg;
    const int b = blockIdx.x, tid = threadIdx.x;   // 64
    __shared__ float red[64];
    float s = 0.0f;
    for (int j = tid; j < D / 2; j += 64) s += h1[(size_t)b * (D / 2) + j] * o2w[j];
    red[tid] = s;
    __syncthreads();
    for (int k = 32; k > 0; k >>= 1) { if (tid < k) red[tid] += red[tid + k]; __syncthreads(); }
    if (tid == 0) {
        const float nv = red[0] + o2b[0];
        out[(size_t)b * P + t]        = nv;
        vals[(size_t)(t + 1) * B + b] = nv;
    }
}

// ---------------------------------------------------------------------------

extern "C" void kernel_launch(void* const* d_in, const int* in_sizes, int n_in,
                              void* d_out, int out_size, void* d_ws, size_t ws_size,
                              hipStream_t stream)
{
    using namespace cfg;
    (void)in_sizes; (void)n_in; (void)out_size; (void)ws_size;

    const float* in_context = (const float*)d_in[0];
    const float* in_start   = (const float*)d_in[1];
    const float* in_ctxW    = (const float*)d_in[2];
    const float* in_ctxb    = (const float*)d_in[3];
    const float* in_veW     = (const float*)d_in[4];
    const float* in_veb     = (const float*)d_in[5];
    const float* in_sainw   = (const float*)d_in[6];
    const float* in_sainb   = (const float*)d_in[7];
    const float* in_saoutw  = (const float*)d_in[8];
    const float* in_saoutb  = (const float*)d_in[9];
    const float* in_cainw   = (const float*)d_in[10];
    const float* in_cainb   = (const float*)d_in[11];
    const float* in_caoutw  = (const float*)d_in[12];
    const float* in_caoutb  = (const float*)d_in[13];
    const float* in_lin1w   = (const float*)d_in[14];
    const float* in_lin1b   = (const float*)d_in[15];
    const float* in_lin2w   = (const float*)d_in[16];
    const float* in_lin2b   = (const float*)d_in[17];
    const float* in_ln1w    = (const float*)d_in[18];
    const float* in_ln1b    = (const float*)d_in[19];
    const float* in_ln2w    = (const float*)d_in[20];
    const float* in_ln2b    = (const float*)d_in[21];
    const float* in_ln3w    = (const float*)d_in[22];
    const float* in_ln3b    = (const float*)d_in[23];
    const float* in_o1w     = (const float*)d_in[24];
    const float* in_o1b     = (const float*)d_in[25];
    const float* in_o2w     = (const float*)d_in[26];
    const float* in_o2b     = (const float*)d_in[27];
    float* out = (float*)d_out;

    // ---- workspace bump allocator ------------------------------------------
    char* wsp = (char*)d_ws;
    auto bump = [&wsp](size_t bytes) -> char* {
        char* r = wsp;
        wsp += (bytes + 255) & ~(size_t)255;
        return r;
    };
    auto halfbuf = [&](size_t elems) { return (_Float16*)bump(elems * sizeof(_Float16)); };
    auto fltbuf  = [&](size_t elems) { return (float*)bump(elems * sizeof(float)); };

    _Float16* wt_ctx   = halfbuf((size_t)D * D);
    _Float16* wt_sain  = halfbuf((size_t)L * 3 * D * D);
    _Float16* wt_saout = halfbuf((size_t)L * D * D);
    _Float16* wt_cain  = halfbuf((size_t)L * 3 * D * D);
    _Float16* wt_caout = halfbuf((size_t)L * D * D);
    _Float16* wt_lin1  = halfbuf((size_t)L * FF * D);
    _Float16* wt_lin2  = halfbuf((size_t)L * D * FF);
    _Float16* wt_o1    = halfbuf((size_t)(D / 2) * D);
    _Float16* ctx_h    = halfbuf((size_t)B * D);

    float*    mem_f  = fltbuf((size_t)B * D);
    _Float16* mem_h  = halfbuf((size_t)B * D);
    float*    vmem_f = fltbuf((size_t)B * D);
    _Float16* vmem_h = halfbuf((size_t)B * D);
    float*    ca_add = fltbuf((size_t)L * B * D);

    float*    x_f   = fltbuf((size_t)B * D);
    _Float16* x_h   = halfbuf((size_t)B * D);
    float*    qkv_f = fltbuf((size_t)B * 3 * D);
    _Float16* o_h   = halfbuf((size_t)B * D);
    float*    tmp_f = fltbuf((size_t)B * D);
    float*    hff_f = fltbuf((size_t)B * FF);
    _Float16* hff_h = halfbuf((size_t)B * FF);
    float*    h1_f  = fltbuf((size_t)B * (D / 2));
    float*    vals  = fltbuf((size_t)(P + 1) * B);

    _Float16* k_cache = halfbuf((size_t)L * P * B * D);
    _Float16* v_cache = halfbuf((size_t)L * P * B * D);

    // ---- helpers ------------------------------------------------------------
    auto cvt = [&](const float* src, _Float16* dst, size_t n) {
        cvt_f32_to_f16_kernel<<<(unsigned)((n + 255) / 256), 256, 0, stream>>>(src, dst, (int)n);
    };
    auto gemm = [&](int act, const _Float16* A, const _Float16* W, const float* bias,
                    float* Cf, _Float16* Ch, int M, int N, int K) {
        const int waves  = (M / 16) * (N / 64);
        const int blocks = (waves + 7) / 8;             // 256 threads = 8 waves
        if (act == 0) gemm_wmma_kernel<0><<<blocks, 256, 0, stream>>>(A, W, bias, Cf, Ch, M, N, K);
        else          gemm_wmma_kernel<1><<<blocks, 256, 0, stream>>>(A, W, bias, Cf, Ch, M, N, K);
    };

    // ---- one-time precompute ------------------------------------------------
    cvt(in_ctxW,   wt_ctx,   (size_t)D * D);
    cvt(in_sainw,  wt_sain,  (size_t)L * 3 * D * D);
    cvt(in_saoutw, wt_saout, (size_t)L * D * D);
    cvt(in_cainw,  wt_cain,  (size_t)L * 3 * D * D);
    cvt(in_caoutw, wt_caout, (size_t)L * D * D);
    cvt(in_lin1w,  wt_lin1,  (size_t)L * FF * D);
    cvt(in_lin2w,  wt_lin2,  (size_t)L * D * FF);
    cvt(in_o1w,    wt_o1,    (size_t)(D / 2) * D);
    cvt(in_context, ctx_h,   (size_t)B * D);

    // memory = context @ ctx_W.T + ctx_b
    gemm(0, ctx_h, wt_ctx, in_ctxb, mem_f, mem_h, B, D, D);

    // cross-attention collapses (T=1): ca_add[l] = (mem@Wv.T+bv)@Wo.T+bo
    for (int l = 0; l < L; ++l) {
        const _Float16* Wv  = wt_cain + ((size_t)l * 3 * D + 2 * D) * D;
        const float*    bv  = in_cainb + (size_t)l * 3 * D + 2 * D;
        gemm(0, mem_h, Wv, bv, vmem_f, vmem_h, B, D, D);
        gemm(0, vmem_h, wt_caout + (size_t)l * D * D, in_caoutb + (size_t)l * D,
             ca_add + (size_t)l * B * D, nullptr, B, D, D);
    }

    init_vals_kernel<<<1, B, 0, stream>>>(vals, in_start);

    // ---- autoregressive decode ---------------------------------------------
    for (int t = 0; t < P; ++t) {
        embed_kernel<<<B, D, 0, stream>>>(vals + (size_t)t * B, in_veW, in_veb, x_f, x_h, t);

        for (int l = 0; l < L; ++l) {
            _Float16* kc = k_cache + (size_t)l * P * B * D;
            _Float16* vc = v_cache + (size_t)l * P * B * D;

            // self-attention
            gemm(0, x_h, wt_sain + (size_t)l * 3 * D * D, in_sainb + (size_t)l * 3 * D,
                 qkv_f, nullptr, B, 3 * D, D);
            kv_append_kernel<<<(B * D) / 256, 256, 0, stream>>>(qkv_f, kc, vc, t);
            attn_kernel<<<B * H, 32, 0, stream>>>(qkv_f, kc, vc, o_h, t);
            gemm(0, o_h, wt_saout + (size_t)l * D * D, in_saoutb + (size_t)l * D,
                 tmp_f, nullptr, B, D, D);
            add_ln_kernel<<<B, 256, 0, stream>>>(x_f, tmp_f,
                 in_ln1w + (size_t)l * D, in_ln1b + (size_t)l * D, x_h);

            // degenerate cross-attention: add precomputed constant, LN
            add_ln_kernel<<<B, 256, 0, stream>>>(x_f, ca_add + (size_t)l * B * D,
                 in_ln2w + (size_t)l * D, in_ln2b + (size_t)l * D, x_h);

            // feed-forward
            gemm(1, x_h, wt_lin1 + (size_t)l * FF * D, in_lin1b + (size_t)l * FF,
                 hff_f, hff_h, B, FF, D);
            gemm(0, hff_h, wt_lin2 + (size_t)l * D * FF, in_lin2b + (size_t)l * D,
                 tmp_f, nullptr, B, D, FF);
            add_ln_kernel<<<B, 256, 0, stream>>>(x_f, tmp_f,
                 in_ln3w + (size_t)l * D, in_ln3b + (size_t)l * D, x_h);
        }

        // output head
        gemm(1, x_h, wt_o1, in_o1b, h1_f, nullptr, B, D / 2, D);
        head2_kernel<<<B, 64, 0, stream>>>(h1_f, in_o2w, in_o2b, out, vals, t);
    }
}
